// HydraQuasiseparableMixer_46462956209007
// MI455X (gfx1250) — matrix operand order repaired and checked by reading, not compiled, not run on hardware
//
#include <hip/hip_runtime.h>
#include <stdint.h>

// Problem constants (fixed by the reference)
#define B_    8
#define T_    4096
#define D_    512
#define DI_   1024
#define KCONV 7
#define M_TOT (B_ * T_)   // 32768

typedef __attribute__((ext_vector_type(16))) __bf16 v16bf;
typedef __attribute__((ext_vector_type(8)))  float  v8f;

// ---------- scalar helpers (self-contained bf16, RNE) ----------
__device__ __forceinline__ uint16_t f2b(float f) {
    union { float f; uint32_t u; } c; c.f = f;
    uint32_t u = c.u;
    u += 0x7FFFu + ((u >> 16) & 1u);
    return (uint16_t)(u >> 16);
}
__device__ __forceinline__ float b2f(uint16_t b) {
    union { uint32_t u; float f; } c; c.u = ((uint32_t)b) << 16;
    return c.f;
}
__device__ __forceinline__ float silu_f(float x) {
    return x / (1.f + __expf(-x));
}

union FragBF { uint32_t u[8]; v16bf v; };

// 16-byte async copy global -> LDS (ASYNCcnt-tracked, no VGPR round-trip)
__device__ __forceinline__ void async_cp16(uint32_t lds_addr, uint64_t gaddr) {
    asm volatile("global_load_async_to_lds_b128 %0, %1, off"
                 :: "v"(lds_addr), "v"(gaddr) : "memory");
}

// =====================================================================
// WMMA GEMM:  C(MxN) = act( A(MxK) * W(NxK)^T + bias ),  A/W bf16.
// Block: 256 threads (8 wave32), tile 128(M) x 128(N), BK = 64.
// Double-buffered LDS fed by global_load_async_to_lds_b128; per wave per
// 32-deep sub-step: 2 A-frags x 4 B-frags = 8 WMMAs (16 per stage).
// MODE 0: split-store bf16 -> out0 (cols<DI) / out1 (z)
// MODE 1: +bias, SiLU, shift-right store: row m -> m+1 (drop t=T-1)
// MODE 2: +bias, SiLU, flip(shift) store: s -> T-2-s (drop s=T-1)
// MODE 3: f32 store to out0 (N=512)
// =====================================================================
template<int MODE>
__global__ __launch_bounds__(256) void gemm_kernel(
    const uint16_t* __restrict__ Ap, const uint16_t* __restrict__ W,
    const float* __restrict__ bias, void* __restrict__ out0,
    void* __restrict__ out1, int Kdim)
{
    constexpr int BM = 128, BN = 128, BK = 64;
    constexpr int LDSTR = 72;            // padded row stride (elems), conflict-free
    constexpr int BUFE  = BM * LDSTR;    // elems per buffer (18 KB)
    __shared__ __align__(16) uint16_t As[2 * BUFE];
    __shared__ __align__(16) uint16_t Bs[2 * BUFE];

    const int tid  = threadIdx.x;
    const int lane = tid & 31;
    const int wave = tid >> 5;
    const int h    = lane >> 4;      // lane half
    const int l16  = lane & 15;
    const int wm   = wave >> 1;      // 0..3 -> m offset wm*32
    const int wn   = wave & 1;       // 0..1 -> n offset wn*64

    const long mBlock = (long)blockIdx.x * BM;
    const long nBlock = (long)blockIdx.y * BN;

    const uint32_t asBase = (uint32_t)(uintptr_t)(void*)As;
    const uint32_t bsBase = (uint32_t)(uintptr_t)(void*)Bs;

    // issue the 8 async 16B copies of one K-stage into buffer b
    auto stage = [&](int b, int k0) {
        #pragma unroll
        for (int i = 0; i < 4; ++i) {
            const int chunk = tid + 256 * i;      // 0..1023
            const int row   = chunk >> 3;         // 0..127
            const int seg   = (chunk & 7) << 3;   // 0..56 elems
            const uint32_t loff = (uint32_t)(b * BUFE + row * LDSTR + seg) * 2u;
            const uint64_t ga = (uint64_t)(uintptr_t)
                (Ap + (mBlock + row) * (long)Kdim + k0 + seg);
            const uint64_t gb = (uint64_t)(uintptr_t)
                (W + (nBlock + row) * (long)Kdim + k0 + seg);
            async_cp16(asBase + loff, ga);
            async_cp16(bsBase + loff, gb);
        }
    };

    v8f acc[2][4];
    #pragma unroll
    for (int mi = 0; mi < 2; ++mi)
        #pragma unroll
        for (int ni = 0; ni < 4; ++ni)
            acc[mi][ni] = (v8f){};

    const int nsteps = Kdim / BK;
    stage(0, 0);

    for (int s = 0; s < nsteps; ++s) {
        const int buf = s & 1;
        if (s + 1 < nsteps) {
            stage(buf ^ 1, (s + 1) * BK);
            // async loads complete in order: <=8 outstanding means the
            // current stage (first 8) has landed in LDS.
            asm volatile("s_wait_asynccnt 0x8" ::: "memory");
        } else {
            asm volatile("s_wait_asynccnt 0x0" ::: "memory");
        }
        __syncthreads();   // all waves' LDS writes visible

        const uint16_t* Ab = &As[buf * BUFE];
        const uint16_t* Bb = &Bs[buf * BUFE];
        #pragma unroll
        for (int kb = 0; kb < BK; kb += 32) {
            FragBF fa[2], fb[4];
            #pragma unroll
            for (int mi = 0; mi < 2; ++mi) {
                const uint16_t* base = &Ab[(wm * 32 + mi * 16 + l16) * LDSTR + kb];
                #pragma unroll
                for (int v = 0; v < 8; ++v) {
                    // A 16x32: k = 16*half + 2*v
                    int kk = (h << 4) + (v << 1);
                    fa[mi].u[v] = *(const uint32_t*)&base[kk];
                }
            }
            #pragma unroll
            for (int ni = 0; ni < 4; ++ni) {
                const uint16_t* base = &Bb[(wn * 64 + ni * 16 + l16) * LDSTR + kb];
                #pragma unroll
                for (int v = 0; v < 8; ++v) {
                    // B 32x16: lanes0-15 K=0..15, lanes16-31 K=16..31
                    int kk = (h << 4) + (v << 1);
                    fb[ni].u[v] = *(const uint32_t*)&base[kk];
                }
            }
            #pragma unroll
            for (int mi = 0; mi < 2; ++mi)
                #pragma unroll
                for (int ni = 0; ni < 4; ++ni)
                    acc[mi][ni] = __builtin_amdgcn_wmma_f32_16x16x32_bf16(
                        false, fa[mi].v, false, fb[ni].v, (short)0,
                        acc[mi][ni], false, false);
        }
        __syncthreads();   // done reading buf before it is overwritten
    }

    // ---- epilogue ----
    #pragma unroll
    for (int mi = 0; mi < 2; ++mi) {
        const int mb = (int)mBlock + wm * 32 + mi * 16 + h * 8;  // C: M = r + 8*half
        #pragma unroll
        for (int ni = 0; ni < 4; ++ni) {
            const int n = (int)nBlock + wn * 64 + ni * 16 + l16;
            #pragma unroll
            for (int r = 0; r < 8; ++r) {
                const int m = mb + r;
                float val = acc[mi][ni][r];
                if constexpr (MODE == 0) {
                    uint16_t bv = f2b(val);
                    if (n < DI_) ((uint16_t*)out0)[(long)m * DI_ + n] = bv;
                    else         ((uint16_t*)out1)[(long)m * DI_ + (n - DI_)] = bv;
                } else if constexpr (MODE == 1) {
                    float v2 = silu_f(val + bias[n]);
                    int t = m & (T_ - 1);
                    if (t != T_ - 1)
                        ((uint16_t*)out0)[(long)(m + 1) * DI_ + n] = f2b(v2);
                } else if constexpr (MODE == 2) {
                    float v2 = silu_f(val + bias[n]);
                    int s2 = m & (T_ - 1);
                    if (s2 != T_ - 1) {
                        long dst = (long)(m - s2) + (T_ - 2 - s2);
                        ((uint16_t*)out0)[dst * DI_ + n] = f2b(v2);
                    }
                } else {
                    ((float*)out0)[(long)m * D_ + n] = val;
                }
            }
        }
    }
}

// ---------- f32 -> bf16 conversion ----------
__global__ __launch_bounds__(256) void cvt_f32_bf16(
    const float* __restrict__ s, uint16_t* __restrict__ d, int n)
{
    int i = blockIdx.x * 256 + threadIdx.x;
    if (i < n) d[i] = f2b(s[i]);
}

// ---------- causal depthwise conv (optionally on time-reversed input) ----------
__global__ __launch_bounds__(256) void conv_dw(
    const uint16_t* __restrict__ xin, const float* __restrict__ w,
    const float* __restrict__ bias, uint16_t* __restrict__ out, int reverse)
{
    const long idx = (long)blockIdx.x * 256 + threadIdx.x;
    const int  c = (int)(idx & (DI_ - 1));
    const long m = idx >> 10;             // row = b*T + t
    const int  t = (int)(m & (T_ - 1));
    const long rowbase = m - t;           // b*T
    float acc = bias[c];
    if (!reverse) {
        #pragma unroll
        for (int k = 0; k < KCONV; ++k) {
            int tt = t - (KCONV - 1) + k;
            if (tt >= 0)
                acc += w[c * KCONV + k] * b2f(xin[(rowbase + tt) * DI_ + c]);
        }
    } else {
        // conv over flipped sequence, result indexed by flipped time s=t
        #pragma unroll
        for (int k = 0; k < KCONV; ++k) {
            int j = t - (KCONV - 1) + k;
            if (j >= 0)
                acc += w[c * KCONV + k] * b2f(xin[(rowbase + (T_ - 1 - j)) * DI_ + c]);
        }
    }
    out[idx] = f2b(acc);
}

// ---------- zero the shift-in rows (t=0 of y_fwd, t=T-1 of y_bwd) ----------
__global__ __launch_bounds__(256) void init_edges(
    uint16_t* __restrict__ yf, uint16_t* __restrict__ yb)
{
    const int idx = blockIdx.x * 256 + threadIdx.x;   // 0..B*DI-1
    const int b = idx >> 10;
    const int c = idx & (DI_ - 1);
    yf[((long)b * T_) * DI_ + c] = 0;
    yb[((long)b * T_ + (T_ - 1)) * DI_ + c] = 0;
}

// ---------- combine: y = (yf + yb + xp*diag) * silu(z) * gs ----------
__global__ __launch_bounds__(256) void combine_kernel(
    const uint16_t* __restrict__ yf, const uint16_t* __restrict__ yb,
    const uint16_t* __restrict__ xp, const uint16_t* __restrict__ z,
    const float* __restrict__ diag, const float* __restrict__ gs,
    uint16_t* __restrict__ yg)
{
    const long idx = (long)blockIdx.x * 256 + threadIdx.x;
    const int  c = (int)(idx & (DI_ - 1));
    float v = b2f(yf[idx]) + b2f(yb[idx]) + b2f(xp[idx]) * diag[c];
    v *= silu_f(b2f(z[idx])) * gs[0];
    yg[idx] = f2b(v);
}

// ---------- residual + layernorm ----------
__global__ __launch_bounds__(256) void ln_kernel(
    const float* __restrict__ x, const float* __restrict__ y,
    const float* __restrict__ g, const float* __restrict__ be,
    float* __restrict__ out)
{
    __shared__ float s1[256];
    __shared__ float s2[256];
    const long m    = blockIdx.x;
    const int  tid  = threadIdx.x;
    const long base = m * D_;
    float h0 = x[base + tid]       + y[base + tid];
    float h1 = x[base + tid + 256] + y[base + tid + 256];
    s1[tid] = h0 + h1;
    s2[tid] = h0 * h0 + h1 * h1;
    __syncthreads();
    for (int s = 128; s > 0; s >>= 1) {
        if (tid < s) { s1[tid] += s1[tid + s]; s2[tid] += s2[tid + s]; }
        __syncthreads();
    }
    float mu  = s1[0] * (1.f / (float)D_);
    float var = s2[0] * (1.f / (float)D_) - mu * mu;
    float rs  = rsqrtf(var + 1e-5f);
    out[base + tid]       = (h0 - mu) * rs * g[tid]       + be[tid];
    out[base + tid + 256] = (h1 - mu) * rs * g[tid + 256] + be[tid + 256];
}

extern "C" void kernel_launch(void* const* d_in, const int* in_sizes, int n_in,
                              void* d_out, int out_size, void* d_ws, size_t ws_size,
                              hipStream_t stream)
{
    (void)in_sizes; (void)n_in; (void)out_size; (void)ws_size;

    const float* x          = (const float*)d_in[0];
    const float* in_proj_w  = (const float*)d_in[1];
    const float* conv_fwd_w = (const float*)d_in[2];
    const float* conv_fwd_b = (const float*)d_in[3];
    const float* proj_fwd_w = (const float*)d_in[4];
    const float* proj_fwd_b = (const float*)d_in[5];
    const float* conv_bwd_w = (const float*)d_in[6];
    const float* conv_bwd_b = (const float*)d_in[7];
    const float* proj_bwd_w = (const float*)d_in[8];
    const float* proj_bwd_b = (const float*)d_in[9];
    const float* diag_w     = (const float*)d_in[10];
    const float* gate_scale = (const float*)d_in[11];
    const float* out_proj_w = (const float*)d_in[12];
    const float* ln_g       = (const float*)d_in[13];
    const float* ln_b       = (const float*)d_in[14];

    char* ws = (char*)d_ws;
    size_t off = 0;
    auto alloc = [&](size_t bytes) -> char* {
        char* p = ws + off;
        off += (bytes + 255) & ~(size_t)255;
        return p;
    };

    uint16_t* wb_in  = (uint16_t*)alloc((size_t)2 * DI_ * D_  * 2); // (2048,512) bf16
    uint16_t* wb_pf  = (uint16_t*)alloc((size_t)DI_ * DI_     * 2);
    uint16_t* wb_pb  = (uint16_t*)alloc((size_t)DI_ * DI_     * 2);
    uint16_t* wb_out = (uint16_t*)alloc((size_t)D_  * DI_     * 2);
    uint16_t* xb     = (uint16_t*)alloc((size_t)M_TOT * D_    * 2); // x in bf16
    uint16_t* xproj  = (uint16_t*)alloc((size_t)M_TOT * DI_   * 2);
    uint16_t* zbuf   = (uint16_t*)alloc((size_t)M_TOT * DI_   * 2);
    uint16_t* convbf = (uint16_t*)alloc((size_t)M_TOT * DI_   * 2); // conv out, later yg
    uint16_t* yfwd   = (uint16_t*)alloc((size_t)M_TOT * DI_   * 2); // later reused as f32 yout
    uint16_t* ybwd   = (uint16_t*)alloc((size_t)M_TOT * DI_   * 2);
    float*    yout   = (float*)yfwd;   // 32768*512*4 == 32768*1024*2 bytes
    uint16_t* yg     = convbf;

    // 1) weights + x -> bf16
    {
        int n1 = 2 * DI_ * D_, n2 = DI_ * DI_, n3 = D_ * DI_, n4 = M_TOT * D_;
        cvt_f32_bf16<<<(n1 + 255) / 256, 256, 0, stream>>>(in_proj_w,  wb_in,  n1);
        cvt_f32_bf16<<<(n2 + 255) / 256, 256, 0, stream>>>(proj_fwd_w, wb_pf,  n2);
        cvt_f32_bf16<<<(n2 + 255) / 256, 256, 0, stream>>>(proj_bwd_w, wb_pb,  n2);
        cvt_f32_bf16<<<(n3 + 255) / 256, 256, 0, stream>>>(out_proj_w, wb_out, n3);
        cvt_f32_bf16<<<(n4 + 255) / 256, 256, 0, stream>>>(x,          xb,     n4);
    }

    // 2) in_proj GEMM: x(32768x512) x W^T(512x2048) -> xproj | z (bf16)
    gemm_kernel<0><<<dim3(M_TOT / 128, (2 * DI_) / 128), 256, 0, stream>>>(
        xb, wb_in, nullptr, xproj, zbuf, D_);

    // 3) zero shift-in rows of yfwd / ybwd
    init_edges<<<(B_ * DI_) / 256, 256, 0, stream>>>(yfwd, ybwd);

    const int NELEM_BLK = (M_TOT * DI_) / 256;   // 131072 blocks

    // 4) fwd depthwise conv, then proj+SiLU GEMM with fused shift-right
    conv_dw<<<NELEM_BLK, 256, 0, stream>>>(xproj, conv_fwd_w, conv_fwd_b, convbf, 0);
    gemm_kernel<1><<<dim3(M_TOT / 128, DI_ / 128), 256, 0, stream>>>(
        convbf, wb_pf, proj_fwd_b, yfwd, nullptr, DI_);

    // 5) bwd (time-reversed) conv, proj+SiLU GEMM with fused flip(shift)
    conv_dw<<<NELEM_BLK, 256, 0, stream>>>(xproj, conv_bwd_w, conv_bwd_b, convbf, 1);
    gemm_kernel<2><<<dim3(M_TOT / 128, DI_ / 128), 256, 0, stream>>>(
        convbf, wb_pb, proj_bwd_b, ybwd, nullptr, DI_);

    // 6) combine + gate -> yg (reuses conv buffer)
    combine_kernel<<<NELEM_BLK, 256, 0, stream>>>(
        yfwd, ybwd, xproj, zbuf, diag_w, gate_scale, yg);

    // 7) out_proj GEMM -> f32 (reuses yfwd storage)
    gemm_kernel<3><<<dim3(M_TOT / 128, D_ / 128), 256, 0, stream>>>(
        yg, wb_out, nullptr, yout, nullptr, DI_);

    // 8) residual + layernorm -> d_out
    ln_kernel<<<M_TOT, 256, 0, stream>>>(x, yout, ln_g, ln_b, (float*)d_out);
}
